// GPTlite_81423989997816
// MI455X (gfx1250) — compile-verified, matrix operand order checked by reading, not run on hardware
//
#include <hip/hip_runtime.h>
#include <hip/hip_bf16.h>

// ---------------------------------------------------------------------------
// GPT-lite forward on gfx1250 (MI455X): bf16 WMMA GEMMs + f32 accumulate.
// Wave tile 32x32 (4 x v_wmma_f32_16x16x32_bf16 per K-step, 2 loads/WMMA).
// ---------------------------------------------------------------------------

typedef __attribute__((ext_vector_type(16))) __bf16 bf16x16;
typedef __attribute__((ext_vector_type(8)))  __bf16 bf16x8;
typedef __attribute__((ext_vector_type(8)))  float  f32x8;

#define GB_B   2
#define GB_S   1024
#define GB_D   1024
#define GB_H   16
#define GB_DH  64
#define GB_FF  4096
#define GB_V   32000
#define GB_L   8
#define GB_BS  (GB_B * GB_S)        // 2048 rows of activations

enum { M_PLAIN = 0, M_QKV = 1, M_VT = 2, M_SCORES = 3, M_ATTNV = 4,
       M_RESID = 5, M_RELU = 6, M_LOGITS = 7 };

// ---------------------------------------------------------------------------
// Store one f32 accumulator element according to the GEMM fusion mode.
// ---------------------------------------------------------------------------
__device__ __forceinline__ void store_one(int mode, void* Cout, float* Resid,
                                          const float* bias, float x,
                                          int row, int col, int N,
                                          long c_bs, int z)
{
    switch (mode) {
    case M_QKV: {   // out[b][h][s][e]  (row = b*S+s, col = h*64+e)
        int b = row >> 10, s = row & 1023;
        int h = col >> 6,  e = col & 63;
        ((__bf16*)Cout)[((((size_t)b * GB_H + h) * GB_S + s) << 6) + e] = (__bf16)x;
        break; }
    case M_VT: {    // out[b][h][e][s]  (V transposed for attn@V)
        int b = row >> 10, s = row & 1023;
        int h = col >> 6,  e = col & 63;
        ((__bf16*)Cout)[(((size_t)b * GB_H + h) * GB_DH + e) * GB_S + s] = (__bf16)x;
        break; }
    case M_SCORES:  // batched bf16 scores (already scaled)
        ((__bf16*)Cout)[(size_t)z * c_bs + (size_t)row * N + col] = (__bf16)x;
        break;
    case M_ATTNV: { // head-concat: out[b][s][h*64+e]   (row = s, col = e, z = b*H+h)
        int b = z >> 4, h = z & 15;
        ((__bf16*)Cout)[((size_t)(b * GB_S + row)) * GB_D + h * GB_DH + col] = (__bf16)x;
        break; }
    case M_RESID: { // h += acc + bias   (f32 in-place residual)
        size_t i = (size_t)row * N + col;
        Resid[i] = Resid[i] + x + bias[col];
        break; }
    case M_RELU: {  // bf16 relu(acc + bias)
        float t = x + bias[col];
        ((__bf16*)Cout)[(size_t)row * N + col] = (__bf16)(t > 0.f ? t : 0.f);
        break; }
    case M_LOGITS:  // f32 acc + bias
        ((float*)Cout)[(size_t)row * N + col] = x + bias[col];
        break;
    default:
        ((__bf16*)Cout)[(size_t)row * N + col] = (__bf16)x;
    }
}

__device__ __forceinline__ bf16x16 load_frag(const __bf16* p)
{
    bf16x8 lo = *(const bf16x8*)(p);
    bf16x8 hi = *(const bf16x8*)(p + 16);
    return __builtin_shufflevector(lo, hi,
        0,1,2,3,4,5,6,7,8,9,10,11,12,13,14,15);
}

// ---------------------------------------------------------------------------
// Generic bf16 WMMA GEMM:  C[M,N] = A[M,K] @ Bt[N,K]^T  (Bt row-major [N][K])
// Block = 256 threads = 8 waves, wave grid 4(M) x 2(N), wave tile 32x32.
// Block tile 128(M) x 64(N).  4 WMMAs per K-step, 8 b128 loads (2 loads/WMMA).
// ---------------------------------------------------------------------------
__global__ __launch_bounds__(256) void gemm_bf16_wmma(
    const __bf16* __restrict__ A, const __bf16* __restrict__ Bt,
    void* __restrict__ Cout, const float* __restrict__ bias,
    float* __restrict__ Resid,
    int M, int N, int K, long a_bs, long b_bs, long c_bs,
    int mode, float scale)
{
    int z = blockIdx.z;
    A  += (size_t)z * a_bs;
    Bt += (size_t)z * b_bs;

    int w  = threadIdx.x >> 5;          // wave id 0..7 (wave32)
    int l  = threadIdx.x & 31;
    int g  = l >> 4;                    // lane group (K-half select)
    int r  = l & 15;                    // row/col within fragment
    int m0 = blockIdx.y * 128 + (w >> 1) * 32;
    int n0 = blockIdx.x * 64  + (w & 1) * 32;
    if (m0 >= M || n0 >= N) return;
    // causal scores: skip wave tiles entirely above the diagonal
    if (mode == M_SCORES && n0 > m0 + 31) return;

    f32x8 acc00 = {}, acc01 = {}, acc10 = {}, acc11 = {};

    const __bf16* ap0 = A  + (size_t)(m0 + r)      * K;
    const __bf16* ap1 = A  + (size_t)(m0 + 16 + r) * K;
    const __bf16* bp0 = Bt + (size_t)(n0 + r)      * K;
    const __bf16* bp1 = Bt + (size_t)(n0 + 16 + r) * K;

    for (int k0 = 0; k0 < K; k0 += 32) {
        int kk = k0 + 8 * g;   // ISA layout: lane group g holds K=8g+0..7, 16+8g+0..7
        bf16x16 a0 = load_frag(ap0 + kk);
        bf16x16 a1 = load_frag(ap1 + kk);
        bf16x16 b0 = load_frag(bp0 + kk);
        bf16x16 b1 = load_frag(bp1 + kk);
        acc00 = __builtin_amdgcn_wmma_f32_16x16x32_bf16(
            false, a0, false, b0, (short)0, acc00, false, false);
        acc01 = __builtin_amdgcn_wmma_f32_16x16x32_bf16(
            false, a0, false, b1, (short)0, acc01, false, false);
        acc10 = __builtin_amdgcn_wmma_f32_16x16x32_bf16(
            false, a1, false, b0, (short)0, acc10, false, false);
        acc11 = __builtin_amdgcn_wmma_f32_16x16x32_bf16(
            false, a1, false, b1, (short)0, acc11, false, false);
    }

    // C/D layout: VGPR v -> row tile_m + v + 8*g, col tile_n + r
#pragma unroll
    for (int v = 0; v < 8; ++v) {
        int ra = m0 + v + 8 * g;
        int rb = ra + 16;
        store_one(mode, Cout, Resid, bias, acc00[v] * scale, ra, n0 + r,      N, c_bs, z);
        store_one(mode, Cout, Resid, bias, acc01[v] * scale, ra, n0 + 16 + r, N, c_bs, z);
        store_one(mode, Cout, Resid, bias, acc10[v] * scale, rb, n0 + r,      N, c_bs, z);
        store_one(mode, Cout, Resid, bias, acc11[v] * scale, rb, n0 + 16 + r, N, c_bs, z);
    }
}

// ---------------------------------------------------------------------------
// fp32 [R][C] -> bf16 [C][R] tiled transpose + convert (LDS 32x32 tile).
// ---------------------------------------------------------------------------
__global__ __launch_bounds__(256) void transpose_cvt_bf16(
    const float* __restrict__ src, __bf16* __restrict__ dst,
    int R, int C, long sbs, long dbs)
{
    __shared__ float tile[32][33];
    src += (size_t)blockIdx.z * sbs;
    dst += (size_t)blockIdx.z * dbs;
    int c0 = blockIdx.x * 32, r0 = blockIdx.y * 32;
    int tx = threadIdx.x & 31, ty = threadIdx.x >> 5;   // 32 x 8
#pragma unroll
    for (int i = 0; i < 32; i += 8)
        tile[ty + i][tx] = src[(size_t)(r0 + ty + i) * C + c0 + tx];
    __syncthreads();
#pragma unroll
    for (int i = 0; i < 32; i += 8)
        dst[(size_t)(c0 + ty + i) * R + r0 + tx] = (__bf16)tile[tx][ty + i];
}

// ---------------------------------------------------------------------------
// h[row] = tok_emb[tok[row]] + pos_emb[row % S]
// ---------------------------------------------------------------------------
__global__ __launch_bounds__(256) void embed_kernel(
    const int* __restrict__ tok, const float* __restrict__ te,
    const float* __restrict__ pe, float* __restrict__ h)
{
    int row = blockIdx.x;
    int t = tok[row];
    int s = row & (GB_S - 1);
    for (int d = threadIdx.x; d < GB_D; d += 256)
        h[(size_t)row * GB_D + d] = te[(size_t)t * GB_D + d] + pe[(size_t)s * GB_D + d];
}

// ---------------------------------------------------------------------------
// LayerNorm f32 row -> bf16 row.  One block (256 threads) per row.
// ---------------------------------------------------------------------------
__global__ __launch_bounds__(256) void layernorm_bf16(
    const float* __restrict__ X, const float* __restrict__ gam,
    const float* __restrict__ bet, __bf16* __restrict__ Y)
{
    __shared__ float red[256];
    int row = blockIdx.x;
    const float* x = X + (size_t)row * GB_D;

    float s = 0.f;
    for (int i = threadIdx.x; i < GB_D; i += 256) s += x[i];
    red[threadIdx.x] = s; __syncthreads();
    for (int o = 128; o > 0; o >>= 1) {
        if (threadIdx.x < o) red[threadIdx.x] += red[threadIdx.x + o];
        __syncthreads();
    }
    float mu = red[0] * (1.f / GB_D);
    __syncthreads();

    float v = 0.f;
    for (int i = threadIdx.x; i < GB_D; i += 256) {
        float d = x[i] - mu; v += d * d;
    }
    red[threadIdx.x] = v; __syncthreads();
    for (int o = 128; o > 0; o >>= 1) {
        if (threadIdx.x < o) red[threadIdx.x] += red[threadIdx.x + o];
        __syncthreads();
    }
    float rs = rsqrtf(red[0] * (1.f / GB_D) + 1e-5f);

    for (int i = threadIdx.x; i < GB_D; i += 256)
        Y[(size_t)row * GB_D + i] = (__bf16)((x[i] - mu) * rs * gam[i] + bet[i]);
}

// ---------------------------------------------------------------------------
// Causal softmax, in-place on bf16 scores; one block per (b,h,m) row.
// Only reads cols 0..m, writes full row (zeros above diagonal).
// ---------------------------------------------------------------------------
__global__ __launch_bounds__(256) void softmax_causal(__bf16* __restrict__ Sc)
{
    __shared__ float red[256];
    size_t rg = blockIdx.x;
    int    m  = (int)(rg & (GB_S - 1));
    size_t bh = rg >> 10;
    __bf16* row = Sc + bh * (size_t)GB_S * GB_S + (size_t)m * GB_S;
    int n = m + 1;

    float mx = -3.0e38f;
    for (int j = threadIdx.x; j < n; j += 256) mx = fmaxf(mx, (float)row[j]);
    red[threadIdx.x] = mx; __syncthreads();
    for (int o = 128; o > 0; o >>= 1) {
        if (threadIdx.x < o) red[threadIdx.x] = fmaxf(red[threadIdx.x], red[threadIdx.x + o]);
        __syncthreads();
    }
    float M = red[0];
    __syncthreads();

    float s = 0.f;
    for (int j = threadIdx.x; j < n; j += 256) s += __expf((float)row[j] - M);
    red[threadIdx.x] = s; __syncthreads();
    for (int o = 128; o > 0; o >>= 1) {
        if (threadIdx.x < o) red[threadIdx.x] += red[threadIdx.x + o];
        __syncthreads();
    }
    float inv = 1.f / red[0];

    for (int j = threadIdx.x; j < GB_S; j += 256) {
        float p = (j <= m) ? __expf((float)row[j] - M) * inv : 0.f;
        row[j] = (__bf16)p;
    }
}

// ---------------------------------------------------------------------------
// Per-row NLL:  nll[r] = logsumexp(logits[r]) - logits[r][target[r]]
// ---------------------------------------------------------------------------
__global__ __launch_bounds__(256) void nll_rows(
    const float* __restrict__ logits, const int* __restrict__ targets,
    float* __restrict__ nll)
{
    __shared__ float red[256];
    int r = blockIdx.x;
    const float* x = logits + (size_t)r * GB_V;

    float mx = -3.0e38f;
    for (int j = threadIdx.x; j < GB_V; j += 256) mx = fmaxf(mx, x[j]);
    red[threadIdx.x] = mx; __syncthreads();
    for (int o = 128; o > 0; o >>= 1) {
        if (threadIdx.x < o) red[threadIdx.x] = fmaxf(red[threadIdx.x], red[threadIdx.x + o]);
        __syncthreads();
    }
    float M = red[0];
    __syncthreads();

    float s = 0.f;
    for (int j = threadIdx.x; j < GB_V; j += 256) s += __expf(x[j] - M);
    red[threadIdx.x] = s; __syncthreads();
    for (int o = 128; o > 0; o >>= 1) {
        if (threadIdx.x < o) red[threadIdx.x] += red[threadIdx.x + o];
        __syncthreads();
    }
    if (threadIdx.x == 0)
        nll[r] = logf(red[0]) + M - x[targets[r]];
}

__global__ __launch_bounds__(256) void mean_reduce(
    const float* __restrict__ nll, float* __restrict__ loss, int n)
{
    __shared__ float red[256];
    float s = 0.f;
    for (int i = threadIdx.x; i < n; i += 256) s += nll[i];
    red[threadIdx.x] = s; __syncthreads();
    for (int o = 128; o > 0; o >>= 1) {
        if (threadIdx.x < o) red[threadIdx.x] += red[threadIdx.x + o];
        __syncthreads();
    }
    if (threadIdx.x == 0) *loss = red[0] / (float)n;
}

// ---------------------------------------------------------------------------
// Host orchestration
// ---------------------------------------------------------------------------
extern "C" void kernel_launch(void* const* d_in, const int* in_sizes, int n_in,
                              void* d_out, int out_size, void* d_ws, size_t ws_size,
                              hipStream_t stream)
{
    const int*   tok    = (const int*)  d_in[0];
    const int*   tgt    = (const int*)  d_in[1];
    const float* te     = (const float*)d_in[2];
    const float* pe     = (const float*)d_in[3];
    const float* Wq     = (const float*)d_in[4];
    const float* Wk     = (const float*)d_in[5];
    const float* Wv     = (const float*)d_in[6];
    const float* Wo     = (const float*)d_in[7];
    const float* bo     = (const float*)d_in[8];
    const float* ln1g   = (const float*)d_in[9];
    const float* ln1b   = (const float*)d_in[10];
    const float* ln2g   = (const float*)d_in[11];
    const float* ln2b   = (const float*)d_in[12];
    const float* W1     = (const float*)d_in[13];
    const float* b1     = (const float*)d_in[14];
    const float* W2     = (const float*)d_in[15];
    const float* b2     = (const float*)d_in[16];
    const float* lnfg   = (const float*)d_in[17];
    const float* lnfb   = (const float*)d_in[18];
    const float* Wout   = (const float*)d_in[19];
    const float* bout   = (const float*)d_in[20];
    (void)in_sizes; (void)n_in; (void)ws_size;

    const size_t MB = 1ull << 20;
    char* W = (char*)d_ws;
    float*  h    = (float*) (W + 0);          //  8 MB  f32 residual stream
    __bf16* xn   = (__bf16*)(W + 8  * MB);    //  4 MB  LN output (reused)
    __bf16* qb   = (__bf16*)(W + 12 * MB);    //  4 MB  Q [B,H,S,DH]
    __bf16* kb   = (__bf16*)(W + 16 * MB);    //  4 MB  K [B,H,S,DH]
    __bf16* vt   = (__bf16*)(W + 20 * MB);    //  4 MB  V [B,H,DH,S]
    __bf16* oc   = (__bf16*)(W + 24 * MB);    //  4 MB  head-concat attn out
    __bf16* ffb  = (__bf16*)(W + 28 * MB);    // 16 MB  relu(ff1)
    __bf16* wq_t = (__bf16*)(W + 44 * MB);    //  2 MB each
    __bf16* wk_t = (__bf16*)(W + 46 * MB);
    __bf16* wv_t = (__bf16*)(W + 48 * MB);
    __bf16* wo_t = (__bf16*)(W + 50 * MB);
    __bf16* w1_t = (__bf16*)(W + 52 * MB);    //  8 MB
    __bf16* w2_t = (__bf16*)(W + 60 * MB);    //  8 MB
    __bf16* sc   = (__bf16*)(W + 68 * MB);    // 64 MB  scores/probs [B*H,S,S]
    __bf16* wo_u = (__bf16*)(W + 132 * MB);   // 62.5MB Wout^T [V,D]
    float*  nllb = (float*) (W + 196 * MB);   //  8 KB

    float* out_f32 = (float*)d_out;
    float* loss    = out_f32 + ((size_t)out_size - 1);

    const dim3 blk(256);
    const float att_scale = 0.03125f;   // D^-0.5 = 1/32 (reference scales by D!)

    // h = tok_emb[tokens] + pos_emb
    embed_kernel<<<GB_BS, blk, 0, stream>>>(tok, te, pe, h);

    for (int l = 0; l < GB_L; ++l) {
        const float* Wq_l = Wq + (size_t)l * GB_H * GB_D * GB_DH;
        const float* Wk_l = Wk + (size_t)l * GB_H * GB_D * GB_DH;
        const float* Wv_l = Wv + (size_t)l * GB_H * GB_D * GB_DH;
        const float* Wo_l = Wo + (size_t)l * GB_D * GB_D;
        const float* W1_l = W1 + (size_t)l * GB_D * GB_FF;
        const float* W2_l = W2 + (size_t)l * GB_FF * GB_D;
        const float* bo_l = bo + (size_t)l * GB_D;
        const float* b1_l = b1 + (size_t)l * GB_FF;
        const float* b2_l = b2 + (size_t)l * GB_D;

        // --- weight convert + transpose to bf16 [N][K] ---
        dim3 gqkv(GB_DH / 32, GB_D / 32, GB_H);   // per-head [D,64] -> [64,D]
        transpose_cvt_bf16<<<gqkv, blk, 0, stream>>>(Wq_l, wq_t, GB_D, GB_DH,
                                                     (long)GB_D * GB_DH, (long)GB_DH * GB_D);
        transpose_cvt_bf16<<<gqkv, blk, 0, stream>>>(Wk_l, wk_t, GB_D, GB_DH,
                                                     (long)GB_D * GB_DH, (long)GB_DH * GB_D);
        transpose_cvt_bf16<<<gqkv, blk, 0, stream>>>(Wv_l, wv_t, GB_D, GB_DH,
                                                     (long)GB_D * GB_DH, (long)GB_DH * GB_D);
        transpose_cvt_bf16<<<dim3(GB_D / 32, GB_D / 32, 1), blk, 0, stream>>>(
            Wo_l, wo_t, GB_D, GB_D, 0, 0);
        transpose_cvt_bf16<<<dim3(GB_FF / 32, GB_D / 32, 1), blk, 0, stream>>>(
            W1_l, w1_t, GB_D, GB_FF, 0, 0);
        transpose_cvt_bf16<<<dim3(GB_D / 32, GB_FF / 32, 1), blk, 0, stream>>>(
            W2_l, w2_t, GB_FF, GB_D, 0, 0);

        // --- xn = LN1(h) ---
        layernorm_bf16<<<GB_BS, blk, 0, stream>>>(h, ln1g + (size_t)l * GB_D,
                                                  ln1b + (size_t)l * GB_D, xn);

        // --- Q, K, V projections (2048 x 1024 x 1024), block tile 128x64 ---
        dim3 gproj(GB_D / 64, GB_BS / 128, 1);
        gemm_bf16_wmma<<<gproj, blk, 0, stream>>>(xn, wq_t, qb, nullptr, nullptr,
            GB_BS, GB_D, GB_D, 0, 0, 0, M_QKV, 1.f);
        gemm_bf16_wmma<<<gproj, blk, 0, stream>>>(xn, wk_t, kb, nullptr, nullptr,
            GB_BS, GB_D, GB_D, 0, 0, 0, M_QKV, 1.f);
        gemm_bf16_wmma<<<gproj, blk, 0, stream>>>(xn, wv_t, vt, nullptr, nullptr,
            GB_BS, GB_D, GB_D, 0, 0, 0, M_VT, 1.f);

        // --- scores = (Q @ K^T) * scale   batched over B*H, causal-skip ---
        dim3 gsc(GB_S / 64, GB_S / 128, GB_B * GB_H);
        gemm_bf16_wmma<<<gsc, blk, 0, stream>>>(qb, kb, sc, nullptr, nullptr,
            GB_S, GB_S, GB_DH,
            (long)GB_S * GB_DH, (long)GB_S * GB_DH, (long)GB_S * GB_S,
            M_SCORES, att_scale);

        // --- causal softmax in-place ---
        softmax_causal<<<GB_B * GB_H * GB_S, blk, 0, stream>>>(sc);

        // --- o = P @ V  (uses V^T layout), store head-concat ---
        dim3 gav(1, GB_S / 128, GB_B * GB_H);
        gemm_bf16_wmma<<<gav, blk, 0, stream>>>(sc, vt, oc, nullptr, nullptr,
            GB_S, GB_DH, GB_S,
            (long)GB_S * GB_S, (long)GB_DH * GB_S, 0, M_ATTNV, 1.f);

        // --- h += o @ Wo + bo ---
        gemm_bf16_wmma<<<gproj, blk, 0, stream>>>(oc, wo_t, nullptr, bo_l, h,
            GB_BS, GB_D, GB_D, 0, 0, 0, M_RESID, 1.f);

        // --- xn = LN2(h) ---
        layernorm_bf16<<<GB_BS, blk, 0, stream>>>(h, ln2g + (size_t)l * GB_D,
                                                  ln2b + (size_t)l * GB_D, xn);

        // --- ff = relu(xn @ W1 + b1) ---
        gemm_bf16_wmma<<<dim3(GB_FF / 64, GB_BS / 128, 1), blk, 0, stream>>>(
            xn, w1_t, ffb, b1_l, nullptr,
            GB_BS, GB_FF, GB_D, 0, 0, 0, M_RELU, 1.f);

        // --- h += ff @ W2 + b2 ---
        gemm_bf16_wmma<<<gproj, blk, 0, stream>>>(ffb, w2_t, nullptr, b2_l, h,
            GB_BS, GB_D, GB_FF, 0, 0, 0, M_RESID, 1.f);
    }

    // --- final LN ---
    layernorm_bf16<<<GB_BS, blk, 0, stream>>>(h, lnfg, lnfb, xn);

    // --- Wout^T convert (once) + logits GEMM into d_out ---
    transpose_cvt_bf16<<<dim3(GB_V / 32, GB_D / 32, 1), blk, 0, stream>>>(
        Wout, wo_u, GB_D, GB_V, 0, 0);
    gemm_bf16_wmma<<<dim3(GB_V / 64, GB_BS / 128, 1), blk, 0, stream>>>(
        xn, wo_u, out_f32, bout, nullptr,
        GB_BS, GB_V, GB_D, 0, 0, 0, M_LOGITS, 1.f);

    // --- loss = mean NLL (deterministic two-stage reduce, no atomics) ---
    nll_rows<<<GB_BS, blk, 0, stream>>>(out_f32, tgt, nllb);
    mean_reduce<<<1, blk, 0, stream>>>(nllb, loss, GB_BS);
}
